// WMSA_G_35519379538389
// MI455X (gfx1250) — matrix operand order repaired and checked by reading, not compiled, not run on hardware
//
#include <hip/hip_runtime.h>
#include <hip/hip_bf16.h>

// ---------- constants ----------
#define NH    6      // heads
#define HD    32     // head dim
#define CH    192    // channels
#define QKVC  576    // 3*CH
#define QKC   384    // 2*CH (q|k region width)
#define P     7      // window size
#define PP    49     // tokens per window
#define HW    64     // windows per side (448/7)
#define NWIN  4096   // HW*HW
#define IMG   448
#define SHIFTV 3
#define SPLIT 4      // P - SHIFT (shift mask boundary)

// LDS layout (bytes)
#define OFF_X    0                        // 64*192 bf16 = 24576 (window x; reused as attn-out)
#define OFF_QK   24576                    // 64*384 bf16 = 49152 (q|k rows, stride 384)
#define OFF_VT   (24576 + 49152)          // 192*64 bf16 = 24576 (v transposed: [c][token])
#define OFF_S    (OFF_VT + 24576)         // 6*64*64 f32 = 98304
#define OFF_P    (OFF_S + 98304)          // 6*64*64 bf16 = 49152
#define LDS_BYTES (OFF_P + 49152)         // 245760 B = 240 KB < 320 KB/WGP

typedef __bf16 bf16_t;
typedef bf16_t v16bf __attribute__((ext_vector_type(16)));
typedef float  v8f   __attribute__((ext_vector_type(8)));

struct alignas(16) U128 { unsigned int a, b, c, d; };
union V16U { v16bf v; unsigned short s[16]; U128 q[2]; };

// Native RNE f32->bf16 (hardware v_cvt on gfx1250; identical rounding to manual RNE)
__device__ __forceinline__ unsigned short f2bf(float f) {
  bf16_t h = (bf16_t)f;
  return __builtin_bit_cast(unsigned short, h);
}

// A-fragment 16x32 bf16: lane holds row (lane&15); K = kb..kb+7 then kb+16..kb+23, kb=(lane>>4)*8
__device__ __forceinline__ v16bf load_frag_A(const unsigned short* base, int row, int rs,
                                             int k0, int lane) {
  int kb = k0 + ((lane >> 4) << 3);
  V16U u;
  u.q[0] = *(const U128*)(base + row * rs + kb);
  u.q[1] = *(const U128*)(base + row * rs + kb + 16);
  return u.v;
}

// B-fragment 32x16 bf16: lane holds col (lane&15); K = kb..kb+7 then kb+16..kb+23, kb=(lane>>4)*8
__device__ __forceinline__ v16bf load_frag_B(const unsigned short* base, int col, int rs,
                                             int k0, int lane) {
  int kb = k0 + ((lane >> 4) << 3);
  V16U u;
  u.q[0] = *(const U128*)(base + col * rs + kb);
  u.q[1] = *(const U128*)(base + col * rs + kb + 16);
  return u.v;
}

// ---------- init kernels ----------
__global__ void cvt_bf16_kernel(const float* __restrict__ in, unsigned short* __restrict__ o, int n) {
  int i = blockIdx.x * 256 + threadIdx.x;
  if (i < n) o[i] = f2bf(in[i]);
}

// bias[h][p][q] = rel_pos[h][pi-qi+6][pj-qj+6] + gaussian_prior(p,q)
__global__ void bias_kernel(const float* __restrict__ rel, float* __restrict__ bias) {
  int i = blockIdx.x * 256 + threadIdx.x;
  if (i >= NH * PP * PP) return;
  int h = i / (PP * PP);
  int rem = i - h * PP * PP;
  int p = rem / PP, q = rem - p * PP;
  int pi = p / P, pj = p - pi * P;
  int qi = q / P, qj = q - qi * P;
  int r0 = pi - qi + P - 1, r1 = pj - qj + P - 1;
  float d2 = (float)((pi - qi) * (pi - qi) + (pj - qj) * (pj - qj));
  const float sig = (float)P / 3.0f;
  float g = __expf(-d2 / (2.0f * sig * sig));
  bias[i] = rel[(h * 13 + r0) * 13 + r1] + g;
}

// ---------- fused per-window kernel ----------
__global__ __launch_bounds__(256)
void wmsa_fused(const float* __restrict__ x,
                const float* __restrict__ b_qkv, const float* __restrict__ b_out,
                const unsigned short* __restrict__ wq, const unsigned short* __restrict__ wo,
                const float* __restrict__ bias, float* __restrict__ out)
{
  extern __shared__ char smem[];
  unsigned short* xs  = (unsigned short*)(smem + OFF_X);   // window x (bf16), later attn-out
  unsigned short* qks = (unsigned short*)(smem + OFF_QK);  // 64x384 bf16 q|k
  unsigned short* vt  = (unsigned short*)(smem + OFF_VT);  // 192x64 bf16 v^T
  float*          ss  = (float*)(smem + OFF_S);            // 6x64x64 scores f32
  unsigned short* ps  = (unsigned short*)(smem + OFF_P);   // 6x64x64 probs bf16

  const int tid  = threadIdx.x;
  // SGPR-resident wave id: tile loops & tile-level branches become scalar control flow,
  // guaranteeing EXEC stays all-ones around every WMMA (ISA requirement).
  const int wave = __builtin_amdgcn_readfirstlane(tid >> 5);
  const int lane = tid & 31;
  const int win  = blockIdx.x;
  const int wh   = win >> 6;    // window row 0..63
  const int wwi  = win & 63;    // window col 0..63
  const bool lastRow = (wh == HW - 1);
  const bool lastCol = (wwi == HW - 1);

  // ---- Phase 1: gather rolled window (roll -3,-3) into LDS bf16, vectorized x4 ----
  for (int e = tid; e < 64 * (CH / 4); e += 256) {         // 64 rows x 48 float4 chunks
    int m = e / (CH / 4), c4 = (e - m * (CH / 4)) * 4;
    unsigned int lo = 0, hi = 0;
    if (m < PP) {
      int ph = m / P, pw = m - ph * P;
      int gh = (wh * P + ph + SHIFTV) % IMG;
      int gw = (wwi * P + pw + SHIFTV) % IMG;
      const float4 f = *(const float4*)(x + (gh * IMG + gw) * CH + c4);
      lo = (unsigned int)f2bf(f.x) | ((unsigned int)f2bf(f.y) << 16);
      hi = (unsigned int)f2bf(f.z) | ((unsigned int)f2bf(f.w) << 16);
    }
    unsigned int* dst = (unsigned int*)(xs + m * CH + c4);
    dst[0] = lo;
    dst[1] = hi;
  }
  __syncthreads();

  // ---- Phase 2: QKV = Xw * Wqkv^T + b ; q scaled; q,k row-major; v stored transposed ----
  for (int t = wave; t < 4 * 36; t += 8) {                 // 4 M-tiles x 36 N-tiles
    int mt = t & 3, nt = t >> 2;
    v8f acc = {};
    #pragma unroll
    for (int ks = 0; ks < 6; ++ks) {                       // K = 192
      v16bf a = load_frag_A(xs, mt * 16 + (lane & 15), CH, ks * 32, lane);
      v16bf b = load_frag_B(wq, nt * 16 + (lane & 15), CH, ks * 32, lane);
      acc = __builtin_amdgcn_wmma_f32_16x16x32_bf16(false, a, false, b, (short)0, acc, false, false);
    }
    int gn = nt * 16 + (lane & 15);
    float bb = b_qkv[gn];
    int mrow = mt * 16 + ((lane >> 4) << 3);
    if (nt < 24) {                                         // q|k tile (scalar branch: nt uniform)
      float sc = (gn < CH) ? 0.17677669529663687f : 1.0f;  // fold softmax scale into q
      #pragma unroll
      for (int i = 0; i < 8; ++i)
        qks[(mrow + i) * QKC + gn] = f2bf((acc[i] + bb) * sc);
    } else {                                               // v tile -> transposed store
      int c = gn - 2 * CH;
      #pragma unroll
      for (int i = 0; i < 8; ++i)                          // contiguous along tokens
        vt[c * 64 + mrow + i] = f2bf(acc[i] + bb);
    }
  }
  __syncthreads();

  // ---- Phase 3: S = q * k^T + bias + shift-mask ----
  for (int t = wave; t < NH * 16; t += 8) {                // 6 heads x (4x4) tiles
    int h = t >> 4, r = t & 15;
    int mt = r >> 2, nt = r & 3;
    v16bf a = load_frag_A(qks, mt * 16 + (lane & 15), QKC, h * HD, lane);
    v16bf b = load_frag_B(qks, nt * 16 + (lane & 15), QKC, CH + h * HD, lane); // k^T
    v8f acc = {};
    acc = __builtin_amdgcn_wmma_f32_16x16x32_bf16(false, a, false, b, (short)0, acc, false, false);
    int n = nt * 16 + (lane & 15);
    int mrow = mt * 16 + ((lane >> 4) << 3);
    #pragma unroll
    for (int i = 0; i < 8; ++i) {
      int m = mrow + i;
      float v = acc[i];
      if (m < PP && n < PP) {
        v += bias[(h * PP + m) * PP + n];
        int pi = m / P, pj = m - pi * P;
        int qi = n / P, qj = n - qi * P;
        if (lastRow && ((pi < SPLIT) != (qi < SPLIT))) v = -1e30f;
        if (lastCol && ((pj < SPLIT) != (qj < SPLIT))) v = -1e30f;
      } else {
        v = -1e30f;
      }
      ss[(h * 64 + m) * 64 + n] = v;
    }
  }
  __syncthreads();

  // ---- Phase 4: softmax rows, write probs as bf16 (padded K cols zeroed) ----
  for (int r = tid; r < NH * 64; r += 256) {
    int m = r & 63;
    float* row = ss + r * 64;
    unsigned short* prow = ps + r * 64;
    if (m < PP) {
      float mx = -1e30f;
      for (int n = 0; n < PP; ++n) mx = fmaxf(mx, row[n]);
      float sum = 0.f;
      for (int n = 0; n < PP; ++n) sum += __expf(row[n] - mx);
      float inv = 1.0f / sum;
      for (int n = 0; n < PP; ++n) prow[n] = f2bf(__expf(row[n] - mx) * inv);
      for (int n = PP; n < 64; ++n) prow[n] = 0;
    } else {
      for (int n = 0; n < 64; ++n) prow[n] = 0;
    }
  }
  __syncthreads();

  // ---- Phase 5: O_h = probs_h * v_h (v^T gives contiguous B-frags); store into xs ----
  for (int t = wave; t < NH * 8; t += 8) {                 // 6 heads x (4 M x 2 N) tiles
    int h = t >> 3, r = t & 7;
    int mt = r >> 1, nt = r & 1;
    int cg = h * HD + nt * 16 + (lane & 15);               // global v channel 0..191
    v8f acc = {};
    #pragma unroll
    for (int ks = 0; ks < 2; ++ks) {                       // K = 64 (padded tokens)
      v16bf a = load_frag_A(ps + h * 64 * 64, mt * 16 + (lane & 15), 64, ks * 32, lane);
      v16bf b = load_frag_B(vt, cg, 64, ks * 32, lane);    // B[k=q][n=c] = vT[c][q]
      acc = __builtin_amdgcn_wmma_f32_16x16x32_bf16(false, a, false, b, (short)0, acc, false, false);
    }
    int mrow = mt * 16 + ((lane >> 4) << 3);
    #pragma unroll
    for (int i = 0; i < 8; ++i)
      xs[(mrow + i) * CH + cg] = f2bf(acc[i]);
  }
  __syncthreads();

  // ---- Phase 6: out = O * Wout^T + b_out, scatter with space-to-depth + roll(+3,+3) ----
  const int rr  = wh >> 4;    // RATIO index -> channel block
  const int w1i = wh & 15;    // row-group within output
  for (int t = wave; t < 4 * 12; t += 8) {                 // 4 M-tiles x 12 N-tiles
    int mt = t / 12, nt = t - mt * 12;
    v8f acc = {};
    #pragma unroll
    for (int ks = 0; ks < 6; ++ks) {
      v16bf a = load_frag_A(xs, mt * 16 + (lane & 15), CH, ks * 32, lane);
      v16bf b = load_frag_B(wo, nt * 16 + (lane & 15), CH, ks * 32, lane);
      acc = __builtin_amdgcn_wmma_f32_16x16x32_bf16(false, a, false, b, (short)0, acc, false, false);
    }
    int n = nt * 16 + (lane & 15);
    float bb = b_out[n];
    int mrow = mt * 16 + ((lane >> 4) << 3);
    int oc = rr * CH + n;                                  // channel 0..767
    #pragma unroll
    for (int i = 0; i < 8; ++i) {
      int m = mrow + i;
      if (m < PP) {
        int ph = m / P, pw = m - ph * P;
        int orow = (w1i * P + ph + SHIFTV) % 112;
        int ocol = (wwi * P + pw + SHIFTV) % IMG;
        out[(orow * IMG + ocol) * (4 * CH) + oc] = acc[i] + bb;
      }
    }
  }
}

// ---------- launch ----------
extern "C" void kernel_launch(void* const* d_in, const int* in_sizes, int n_in,
                              void* d_out, int out_size, void* d_ws, size_t ws_size,
                              hipStream_t stream) {
  (void)in_sizes; (void)n_in; (void)out_size; (void)ws_size;
  const float* x     = (const float*)d_in[0];
  const float* W_qkv = (const float*)d_in[1];
  const float* b_qkv = (const float*)d_in[2];
  const float* rel   = (const float*)d_in[3];
  const float* W_out = (const float*)d_in[4];
  const float* b_out = (const float*)d_in[5];
  float* out = (float*)d_out;

  unsigned short* wq  = (unsigned short*)d_ws;             // 576*192 bf16
  unsigned short* wo  = wq + QKVC * CH;                    // 192*192 bf16
  float* bias         = (float*)(wo + CH * CH);            // 6*49*49 f32

  cvt_bf16_kernel<<<(QKVC * CH + 255) / 256, 256, 0, stream>>>(W_qkv, wq, QKVC * CH);
  cvt_bf16_kernel<<<(CH * CH + 255) / 256, 256, 0, stream>>>(W_out, wo, CH * CH);
  bias_kernel<<<(NH * PP * PP + 255) / 256, 256, 0, stream>>>(rel, bias);

  wmsa_fused<<<NWIN, 256, LDS_BYTES, stream>>>(x, b_qkv, b_out, wq, wo, bias, out);
}
